// ContourGeneratorRNN_61340722921892
// MI455X (gfx1250) — matrix operand order repaired and checked by reading, not compiled, not run on hardware
//
#include <hip/hip_runtime.h>

// ---------------------------------------------------------------------------
// GRU (T=512, B=64, F=128, H=512, O=16) for gfx1250 (MI455X), wave32 + WMMA.
//
//  1) init_kernel: zero the cross-WG sync counter.
//  2) gru_scan_kernel: 16 persistent workgroups (6 waves each). WG g owns
//     h-columns [32g,32g+32) and the matching 96 gate columns (r,z,n).
//     w_hh/w_ih slices are converted fp32->f16 once into LDS and stay
//     resident for all 512 steps. Each step:
//        acc_x = x_t @ w_ih_slice^T   (WMMA f16, K=128)
//        acc_h = h   @ w_hh_slice^T   (WMMA f16, K=512)
//        gates + h_new (VALU, f32), h_new -> hs[t] (f16, global)
//        atomic arrive/spin barrier across the 16 WGs
//        reload full h (= hs[t]) via GLOBAL_LOAD_ASYNC_TO_LDS_B128 (ASYNCcnt)
//  3) out_proj_kernel: y = hs @ w_out^T + b_out via WMMA (M=32768,N=16,K=512)
// ---------------------------------------------------------------------------

typedef __attribute__((ext_vector_type(16))) _Float16 v16h;
typedef __attribute__((ext_vector_type(8)))  _Float16 v8h;
typedef __attribute__((ext_vector_type(8)))  float    v8f;

#define T_STEPS 512
#define B_DIM   64
#define F_DIM   128
#define H_DIM   512
#define O_DIM   16

#define NWG          16    // scan workgroups (one per WGP)
#define SCAN_THREADS 192   // 6 waves of 32
#define MT           4     // M tiles (64 rows / 16)
// each wave owns one of 6 N-tiles (96 gate columns / 16)

// LDS strides in f16 elements (padded, keep 16B alignment per row)
#define WHH_LS 520
#define WIH_LS 136
#define H_LS   520
#define X_LS   136

// LDS budget (bytes):
//  sWhh 96*520*2 + sWih 96*136*2 + sH 64*520*2 + sX 64*136*2  = 209920
//  sGx/sGh 2*64*96*4 + sBih/sBhh 2*1536*4                     =  61440
#define SMEM_BYTES (209920 + 61440)

__device__ __forceinline__ v8f vzero8() {
  v8f v;
#pragma unroll
  for (int i = 0; i < 8; ++i) v[i] = 0.f;
  return v;
}

// Load a 16-half WMMA fragment as two 16-byte chunks at element offsets kA, kB.
__device__ __forceinline__ v16h frag_ld(const _Float16* p, int kA, int kB) {
  v8h c0 = *(const v8h*)(p + kA);
  v8h c1 = *(const v8h*)(p + kB);
  return __builtin_shufflevector(c0, c1, 0, 1, 2, 3, 4, 5, 6, 7,
                                 8, 9, 10, 11, 12, 13, 14, 15);
}

// Raw 32-bit LDS byte offset of a generic pointer that points into LDS.
__device__ __forceinline__ unsigned lds_off(const void* p) {
  return (unsigned)(unsigned long long)
      (__attribute__((address_space(3))) const char*)p;
}

__global__ void init_kernel(unsigned* ctr) { ctr[0] = 0u; }

__global__ void __launch_bounds__(SCAN_THREADS, 1)
gru_scan_kernel(const float* __restrict__ x,      // (T,B,F)
                const float* __restrict__ w_ih,   // (3H,F)  gate order r,z,n
                const float* __restrict__ w_hh,   // (3H,H)
                const float* __restrict__ b_ih,   // (3H)
                const float* __restrict__ b_hh,   // (3H)
                _Float16*    __restrict__ hs,     // (T,B,H) f16 out / h exchange
                unsigned*    __restrict__ ctr)
{
  extern __shared__ __align__(16) char smem[];
  _Float16* sWhh = (_Float16*)smem;            // [96][WHH_LS]
  _Float16* sWih = sWhh + 96 * WHH_LS;         // [96][WIH_LS]
  _Float16* sH   = sWih + 96 * WIH_LS;         // [64][H_LS]   h_{t-1}
  _Float16* sX   = sH   + 64 * H_LS;           // [64][X_LS]   x_t
  float*    sGx  = (float*)(sX + 64 * X_LS);   // [64][96] x-side preacts
  float*    sGh  = sGx + 64 * 96;              // [64][96] h-side preacts
  float*    sBih = sGh + 64 * 96;              // [1536]
  float*    sBhh = sBih + 3 * H_DIM;           // [1536]

  const int g    = blockIdx.x;       // 0..15: column-group owner
  const int tid  = threadIdx.x;
  const int lane = tid & 31;
  const int wv   = tid >> 5;         // 0..5: N-tile of this wave
  const int idx  = lane & 15;
  const int hsel = (lane >> 4) & 1;  // 0 = lanes 0-15, 1 = lanes 16-31

  const unsigned sH_base = lds_off(sH);

  // ---- one-time setup: resident f16 weight slices, biases, h0=0, x0 ----
  for (int i = tid; i < 96 * H_DIM; i += SCAN_THREADS) {
    int c = i >> 9, k = i & (H_DIM - 1);
    int grow = (c >> 5) * H_DIM + g * 32 + (c & 31);     // global gate row
    sWhh[c * WHH_LS + k] = (_Float16)w_hh[grow * H_DIM + k];
  }
  for (int i = tid; i < 96 * F_DIM; i += SCAN_THREADS) {
    int c = i >> 7, k = i & (F_DIM - 1);
    int grow = (c >> 5) * H_DIM + g * 32 + (c & 31);
    sWih[c * WIH_LS + k] = (_Float16)w_ih[grow * F_DIM + k];
  }
  for (int i = tid; i < 3 * H_DIM; i += SCAN_THREADS) {
    sBih[i] = b_ih[i];
    sBhh[i] = b_hh[i];
  }
  for (int i = tid; i < B_DIM * H_DIM; i += SCAN_THREADS)
    sH[(i >> 9) * H_LS + (i & (H_DIM - 1))] = (_Float16)0.f;
  for (int i = tid; i < B_DIM * F_DIM; i += SCAN_THREADS)
    sX[(i >> 7) * X_LS + (i & (F_DIM - 1))] = (_Float16)x[i];
  __syncthreads();

  const _Float16* wih_p = sWih + (16 * wv + idx) * WIH_LS;  // B-frag row base
  const _Float16* whh_p = sWhh + (16 * wv + idx) * WHH_LS;

  for (int t = 0; t < T_STEPS; ++t) {
    // prefetch next x tile while we compute
    if (t + 1 < T_STEPS && tid < B_DIM) {
      const float* nx = x + (size_t)(t + 1) * B_DIM * F_DIM;
      __builtin_prefetch(nx + tid * F_DIM, 0, 1);
    }

    v8f accX[MT], accH[MT];
#pragma unroll
    for (int m = 0; m < MT; ++m) { accX[m] = vzero8(); accH[m] = vzero8(); }

    // ---- x-side: 64x96 += x_t(64x128) @ w_ih_slice^T, K=128 ----
#pragma unroll
    for (int ks = 0; ks < F_DIM / 32; ++ks) {
      v16h bfr = frag_ld(wih_p, 32 * ks + 16 * hsel, 32 * ks + 16 * hsel + 8);
#pragma unroll
      for (int m = 0; m < MT; ++m) {
        v16h afr = frag_ld(sX + (16 * m + idx) * X_LS,
                           32 * ks + 16 * hsel, 32 * ks + 16 * hsel + 8);
        accX[m] = __builtin_amdgcn_wmma_f32_16x16x32_f16(
            false, afr, false, bfr, (short)0, accX[m], false, false);
      }
    }
    // ---- h-side: 64x96 += h(64x512) @ w_hh_slice^T, K=512 ----
#pragma unroll
    for (int ks = 0; ks < H_DIM / 32; ++ks) {
      v16h bfr = frag_ld(whh_p, 32 * ks + 16 * hsel, 32 * ks + 16 * hsel + 8);
#pragma unroll
      for (int m = 0; m < MT; ++m) {
        v16h afr = frag_ld(sH + (16 * m + idx) * H_LS,
                           32 * ks + 16 * hsel, 32 * ks + 16 * hsel + 8);
        accH[m] = __builtin_amdgcn_wmma_f32_16x16x32_f16(
            false, afr, false, bfr, (short)0, accH[m], false, false);
      }
    }

    // ---- spill pre-activations to LDS (WMMA D layout) ----
#pragma unroll
    for (int m = 0; m < MT; ++m) {
      int row0 = 16 * m + 8 * hsel;
      int col  = 16 * wv + idx;
#pragma unroll
      for (int r = 0; r < 8; ++r) {
        sGx[(row0 + r) * 96 + col] = accX[m][r];
        sGh[(row0 + r) * 96 + col] = accH[m][r];
      }
    }
    __syncthreads();

    // ---- gates + h_new for our 32 columns ----
    _Float16* hs_t = hs + (size_t)t * B_DIM * H_DIM;
    for (int i = tid; i < B_DIM * 32; i += SCAN_THREADS) {
      int b = i >> 5, j = i & 31;
      int col = g * 32 + j;
      float pr = sGx[b * 96 + j]      + sGh[b * 96 + j]      + sBih[col]            + sBhh[col];
      float pz = sGx[b * 96 + j + 32] + sGh[b * 96 + j + 32] + sBih[H_DIM + col]    + sBhh[H_DIM + col];
      float pn = sGx[b * 96 + j + 64] + sBih[2 * H_DIM + col];
      float hn = sGh[b * 96 + j + 64] + sBhh[2 * H_DIM + col];
      float r = 1.f / (1.f + __expf(-pr));
      float z = 1.f / (1.f + __expf(-pz));
      float n = tanhf(pn + r * hn);
      float hp = (float)sH[b * H_LS + col];
      hs_t[b * H_DIM + col] = (_Float16)((1.f - z) * n + z * hp);
    }
    if (t == T_STEPS - 1) break;

    __syncthreads();
    __threadfence();
    if (tid == 0) {  // device-scope arrive + spin barrier across the 16 WGs
      __hip_atomic_fetch_add(ctr, 1u, __ATOMIC_RELEASE, __HIP_MEMORY_SCOPE_AGENT);
      unsigned tgt = (unsigned)NWG * (unsigned)(t + 1);
      while (__hip_atomic_load(ctr, __ATOMIC_ACQUIRE, __HIP_MEMORY_SCOPE_AGENT) < tgt)
        __builtin_amdgcn_s_sleep(2);
    }
    __syncthreads();

    // ---- reload full h_t (64x512 f16): async global -> LDS (ASYNCcnt) ----
    {
      unsigned long long hsrc = (unsigned long long)hs_t;  // uniform -> SGPR
      for (int ci = tid; ci < (B_DIM * H_DIM) / 8; ci += SCAN_THREADS) {
        int row = ci >> 6, kc = (ci & 63) * 8;
        unsigned laddr = sH_base + (unsigned)(row * H_LS + kc) * 2u;
        unsigned voff  = (unsigned)(row * H_DIM + kc) * 2u;
        asm volatile("global_load_async_to_lds_b128 %0, %1, %2"
                     :: "v"(laddr), "v"(voff), "s"(hsrc) : "memory");
      }
      asm volatile("s_wait_asynccnt 0x0" ::: "memory");
    }
    // ---- stage x_{t+1} (fp32 -> f16 conversion, regular path) ----
    const float* xn = x + (size_t)(t + 1) * B_DIM * F_DIM;
    for (int i = tid; i < B_DIM * F_DIM; i += SCAN_THREADS)
      sX[(i >> 7) * X_LS + (i & (F_DIM - 1))] = (_Float16)xn[i];
    __syncthreads();
  }
}

// ---- y(32768x16) = hs(32768x512,f16) @ w_out^T + b_out --------------------
#define PROJ_THREADS 256
__global__ void __launch_bounds__(PROJ_THREADS, 2)
out_proj_kernel(const _Float16* __restrict__ hs,
                const float* __restrict__ w_out,   // (O,H)
                const float* __restrict__ b_out,   // (O)
                float* __restrict__ y)             // (T,B,O)
{
  int wid  = (int)((blockIdx.x * PROJ_THREADS + threadIdx.x) >> 5); // 0..2047
  int lane = threadIdx.x & 31;
  int idx  = lane & 15;
  int hsel = (lane >> 4) & 1;

  int m0 = wid * 16;
  const _Float16* arow = hs + (size_t)(m0 + idx) * H_DIM;
  const float*    bcol = w_out + (size_t)idx * H_DIM;  // B col = w_out row o

  v8f acc = vzero8();
#pragma unroll
  for (int ks = 0; ks < H_DIM / 32; ++ks) {
    v16h afr = frag_ld(arow, 32 * ks + 16 * hsel, 32 * ks + 16 * hsel + 8);
    v16h bfr;
    const float* bp = bcol + 32 * ks + 16 * hsel;
#pragma unroll
    for (int i = 0; i < 16; ++i) bfr[i] = (_Float16)bp[i];
    acc = __builtin_amdgcn_wmma_f32_16x16x32_f16(
        false, afr, false, bfr, (short)0, acc, false, false);
  }
  float bo = b_out[idx];
  int row0 = m0 + 8 * hsel;
#pragma unroll
  for (int r = 0; r < 8; ++r)
    y[(size_t)(row0 + r) * O_DIM + idx] = acc[r] + bo;
}

extern "C" void kernel_launch(void* const* d_in, const int* in_sizes, int n_in,
                              void* d_out, int out_size, void* d_ws, size_t ws_size,
                              hipStream_t stream) {
  (void)in_sizes; (void)n_in; (void)out_size; (void)ws_size;
  const float* x     = (const float*)d_in[0];
  const float* w_ih  = (const float*)d_in[1];
  const float* w_hh  = (const float*)d_in[2];
  const float* b_ih  = (const float*)d_in[3];
  const float* b_hh  = (const float*)d_in[4];
  const float* w_out = (const float*)d_in[5];
  const float* b_out = (const float*)d_in[6];

  unsigned* ctr = (unsigned*)d_ws;
  _Float16* hs  = (_Float16*)((char*)d_ws + 256);   // 512*64*512 f16 = 32 MB

  // allow >64KB dynamic LDS (not a stream op; safe during capture)
  hipFuncSetAttribute((const void*)gru_scan_kernel,
                      hipFuncAttributeMaxDynamicSharedMemorySize, SMEM_BYTES);

  init_kernel<<<1, 1, 0, stream>>>(ctr);
  gru_scan_kernel<<<NWG, SCAN_THREADS, SMEM_BYTES, stream>>>(
      x, w_ih, w_hh, b_ih, b_hh, hs, ctr);

  const int rows = T_STEPS * B_DIM;                 // 32768
  const int waves = rows / 16;                      // 2048
  out_proj_kernel<<<waves * 32 / PROJ_THREADS, PROJ_THREADS, 0, stream>>>(
      hs, w_out, b_out, (float*)d_out);
}
